// VQuantizer_57595511439943
// MI455X (gfx1250) — matrix operand order repaired and checked by reading, not compiled, not run on hardware
//
#include <hip/hip_runtime.h>

// ---------------------------------------------------------------------------
// VQ quantizer for MI455X (gfx1250, wave32).
// distances GEMM via v_wmma_f32_16x16x32_bf16; codebook staged to LDS with
// gfx1250 async-copy (global_load_async_to_lds_b128 + s_wait_asynccnt) when
// the toolchain exposes the builtins, else register prefetch + ds_store.
// ---------------------------------------------------------------------------

typedef __attribute__((ext_vector_type(16))) __bf16 v16bf;
typedef __attribute__((ext_vector_type(8)))  float  v8f;

#define D 128           // feature dim
#define NB 64           // codebook rows staged per LDS block (16 KB bf16)
#define PITCH_B 272     // padded LDS row pitch (68 dwords -> bank-balanced)

#if __has_builtin(__builtin_amdgcn_global_load_async_to_lds_b128) && \
    __has_builtin(__builtin_amdgcn_s_wait_asynccnt)
#define USE_ASYNC_LDS 1
#else
#define USE_ASYNC_LDS 0
#endif

#if USE_ASYNC_LDS
// Exact pointee type from clang's prototype: int __vector(4).
typedef int async_v4i __attribute__((vector_size(16)));
typedef __attribute__((address_space(1))) async_v4i* g_v4i_ptr;
typedef __attribute__((address_space(3))) async_v4i* l_v4i_ptr;
#endif

union V16 {
    v16bf v;
    uint4 q[2];
};

__device__ __forceinline__ unsigned pack_bf16x2(float a, float b) {
    unsigned ua = __float_as_uint(a);
    unsigned ub = __float_as_uint(b);
    ua += 0x7fffu + ((ua >> 16) & 1u);   // RNE
    ub += 0x7fffu + ((ub >> 16) & 1u);
    return (ua >> 16) | (ub & 0xffff0000u);
}

// ---------------- Kernel 1: LayerNorm, bf16 cast, ||xn||^2 -----------------
__global__ __launch_bounds__(256) void k_layernorm(
    const float* __restrict__ x, float* __restrict__ xn,
    unsigned short* __restrict__ xnbf, float* __restrict__ x2) {
    const int wave = threadIdx.x >> 5;
    const int lane = threadIdx.x & 31;
    const int row  = blockIdx.x * 8 + wave;

    float4 v = ((const float4*)(x + (size_t)row * D))[lane];
    float s = v.x + v.y + v.z + v.w;
#pragma unroll
    for (int off = 1; off < 32; off <<= 1) s += __shfl_xor(s, off, 32);
    const float mu = s * (1.0f / (float)D);

    float dx = v.x - mu, dy = v.y - mu, dz = v.z - mu, dw = v.w - mu;
    float vs = dx * dx + dy * dy + dz * dz + dw * dw;
#pragma unroll
    for (int off = 1; off < 32; off <<= 1) vs += __shfl_xor(vs, off, 32);
    const float var = vs * (1.0f / (float)D);
    const float rs  = rsqrtf(var + 1e-5f);

    float4 o;
    o.x = dx * rs; o.y = dy * rs; o.z = dz * rs; o.w = dw * rs;
    ((float4*)(xn + (size_t)row * D))[lane] = o;

    uint2 p;
    p.x = pack_bf16x2(o.x, o.y);
    p.y = pack_bf16x2(o.z, o.w);
    ((uint2*)(xnbf + (size_t)row * D))[lane] = p;

    if (lane == 0) x2[row] = (float)D * var * rs * rs;  // == sum(xn^2)
}

// ---------------- Kernel 2: codebook bf16 cast + ||e||^2 -------------------
__global__ __launch_bounds__(256) void k_prepcb(
    const float* __restrict__ cb, unsigned short* __restrict__ cbbf,
    float* __restrict__ e2) {
    const int wave = threadIdx.x >> 5;
    const int lane = threadIdx.x & 31;
    const int row  = blockIdx.x * 8 + wave;

    float4 v = ((const float4*)(cb + (size_t)row * D))[lane];
    float ss = v.x * v.x + v.y * v.y + v.z * v.z + v.w * v.w;
#pragma unroll
    for (int off = 1; off < 32; off <<= 1) ss += __shfl_xor(ss, off, 32);

    uint2 p;
    p.x = pack_bf16x2(v.x, v.y);
    p.y = pack_bf16x2(v.z, v.w);
    ((uint2*)(cbbf + (size_t)row * D))[lane] = p;

    if (lane == 0) e2[row] = ss;
}

// ---------------- Kernel 3: distances via WMMA bf16 + argmin ---------------
__global__ __launch_bounds__(256) void k_argmin(
    const unsigned short* __restrict__ xnbf, const float* __restrict__ x2,
    const unsigned short* __restrict__ cbbf, const float* __restrict__ e2,
    int* __restrict__ ind_i32, float* __restrict__ ind_f32, int K) {
    __shared__ __align__(16) char smem[2 * NB * PITCH_B];

    const int tid  = threadIdx.x;
    const int wave = tid >> 5;
    const int lane = tid & 31;
    const int half = lane >> 4;   // 0: lanes 0-15, 1: lanes 16-31
    const int lq   = lane & 15;

    const int mrow = (blockIdx.x * 8 + wave) * 16;

    // ---- A matrix: 16x128 bf16, 4 chunks of 16x32, resident in VGPRs.
    V16 A[4];
    {
        const unsigned short* arow = xnbf + (size_t)(mrow + lq) * D;
#pragma unroll
        for (int c = 0; c < 4; ++c) {
            const int kb = half * 8 + c * 32;
            A[c].q[0] = *(const uint4*)(arow + kb);
            A[c].q[1] = *(const uint4*)(arow + kb + 16);
        }
    }

    // C/D layout: VGPR r -> M = r + 8*half, lane -> N = lq.
    float x2n[8];
#pragma unroll
    for (int r = 0; r < 8; ++r) x2n[r] = -0.5f * x2[mrow + half * 8 + r];

    float bv[8];
    int   bn[8];
#pragma unroll
    for (int r = 0; r < 8; ++r) { bv[r] = -3.4e38f; bn[r] = 0; }

    const int nblocks = K / NB;

#if USE_ASYNC_LDS
    // ---- gfx1250 async copy: global -> LDS, tracked by ASYNCcnt -----------
    auto issueB = [&](int b, int bufi) {
        const char* src = (const char*)(cbbf + (size_t)b * NB * D);
        char*       dst = smem + bufi * (NB * PITCH_B);
#pragma unroll
        for (int j = 0; j < 4; ++j) {
            const int idx = tid + 256 * j;
            __builtin_amdgcn_global_load_async_to_lds_b128(
                (g_v4i_ptr)(unsigned long long)(src + idx * 16),
                (l_v4i_ptr)(unsigned long long)
                    (dst + (idx >> 4) * PITCH_B + (idx & 15) * 16),
                0, 0);
        }
    };
    issueB(0, 0);
#else
    uint4 pf[4];
    auto fetchB = [&](int b) {
        const uint4* src = (const uint4*)(cbbf + (size_t)b * NB * D);
#pragma unroll
        for (int j = 0; j < 4; ++j) pf[j] = src[tid + 256 * j];
    };
    auto storeB = [&](int bufi) {
        char* dst = smem + bufi * (NB * PITCH_B);
#pragma unroll
        for (int j = 0; j < 4; ++j) {
            const int idx = tid + 256 * j;
            *(uint4*)(dst + (idx >> 4) * PITCH_B + (idx & 15) * 16) = pf[j];
        }
    };
    fetchB(0);
    storeB(0);
    __syncthreads();
#endif

    int buf = 0;
    for (int b = 0; b < nblocks; ++b) {
#if USE_ASYNC_LDS
        if (b + 1 < nblocks) {
            issueB(b + 1, buf ^ 1);                 // fill other buffer
            __builtin_amdgcn_s_wait_asynccnt(4);    // block b's 4 ops done
        } else {
            __builtin_amdgcn_s_wait_asynccnt(0);
        }
        __syncthreads();                            // all waves' copies visible
#else
        if (b + 1 < nblocks) fetchB(b + 1);
#endif

        const char* Bbase = smem + buf * (NB * PITCH_B);
        const int   nbase = b * NB;

#pragma unroll
        for (int sub = 0; sub < NB / 16; ++sub) {
            const int nloc = sub * 16 + lq;
            const int ng   = nbase + nloc;
            const float e2v = e2[ng];

            // Load all 4 B chunks into independent regs so ds_loads of the
            // next subtile overlap with this subtile's WMMAs.
            V16 Bm[4];
            const char* brow = Bbase + nloc * PITCH_B + half * 16;
#pragma unroll
            for (int c = 0; c < 4; ++c) {
                Bm[c].q[0] = *(const uint4*)(brow + c * 64);
                Bm[c].q[1] = *(const uint4*)(brow + c * 64 + 32);
            }

            // acc = -0.5*(||x||^2 + ||e||^2); after dots: acc = x.e - 0.5(..)
            v8f acc;
#pragma unroll
            for (int r = 0; r < 8; ++r) acc[r] = fmaf(-0.5f, e2v, x2n[r]);

#pragma unroll
            for (int c = 0; c < 4; ++c) {
                acc = __builtin_amdgcn_wmma_f32_16x16x32_bf16(
                    false, A[c].v, false, Bm[c].v, (short)0, acc, false, false);
            }

#pragma unroll
            for (int r = 0; r < 8; ++r) {
                if (acc[r] > bv[r]) { bv[r] = acc[r]; bn[r] = ng; }
            }
        }

        __syncthreads();    // all reads of buf done before it is refilled
#if !USE_ASYNC_LDS
        if (b + 1 < nblocks) {
            storeB(buf ^ 1);
            __syncthreads();
        }
#endif
        buf ^= 1;
    }

    // cross-lane argmax reduction within each 16-lane half (argmin of dist)
#pragma unroll
    for (int r = 0; r < 8; ++r) {
        float v = bv[r];
        int   n = bn[r];
#pragma unroll
        for (int off = 1; off < 16; off <<= 1) {
            float ov = __shfl_xor(v, off, 32);
            int   on = __shfl_xor(n, off, 32);
            if (ov > v || (ov == v && on < n)) { v = ov; n = on; }
        }
        if (lq == 0) {
            const int m = mrow + half * 8 + r;
            ind_i32[m] = n;
            ind_f32[m] = (float)n;
        }
    }
}

// ---------------- Kernel 4: gather, straight-through out, loss -------------
__global__ __launch_bounds__(256) void k_gather(
    const float* __restrict__ xn, const float* __restrict__ cb,
    const int* __restrict__ ind, float* __restrict__ qout,
    float* __restrict__ loss, float inv_count) {
    const int wave = threadIdx.x >> 5;
    const int lane = threadIdx.x & 31;
    const int row  = blockIdx.x * 8 + wave;

    const int e = ind[row];
    float4 c  = ((const float4*)(cb + (size_t)e * D))[lane];
    float4 xv = ((const float4*)(xn + (size_t)row * D))[lane];

    float4 dlt, q;
    dlt.x = c.x - xv.x; dlt.y = c.y - xv.y; dlt.z = c.z - xv.z; dlt.w = c.w - xv.w;
    q.x = xv.x + dlt.x; q.y = xv.y + dlt.y; q.z = xv.z + dlt.z; q.w = xv.w + dlt.w;
    ((float4*)(qout + (size_t)row * D))[lane] = q;

    float s = dlt.x * dlt.x + dlt.y * dlt.y + dlt.z * dlt.z + dlt.w * dlt.w;
#pragma unroll
    for (int off = 1; off < 32; off <<= 1) s += __shfl_xor(s, off, 32);
    if (lane == 0) {
        __hip_atomic_fetch_add(loss, s * inv_count, __ATOMIC_RELAXED,
                               __HIP_MEMORY_SCOPE_AGENT);
    }
}

__global__ void k_zero(float* p) {
    if (threadIdx.x == 0 && blockIdx.x == 0) *p = 0.0f;
}

// ---------------------------------------------------------------------------
extern "C" void kernel_launch(void* const* d_in, const int* in_sizes, int n_in,
                              void* d_out, int out_size, void* d_ws, size_t ws_size,
                              hipStream_t stream) {
    const float* x  = (const float*)d_in[0];   // [4,4096,128] f32
    const float* cb = (const float*)d_in[1];   // [8192,128]   f32

    const int BN = in_sizes[0] / D;   // 16384 rows
    const int K  = in_sizes[1] / D;   // 8192 codes

    float* out = (float*)d_out;
    const size_t q_off    = 0;
    const size_t ind_off  = (size_t)BN * D;        // 2097152
    const size_t loss_off = ind_off + (size_t)BN;  // 2113536

    // workspace layout (256B aligned slices)
    char* ws = (char*)d_ws;
    size_t off = 0;
    auto take = [&](size_t bytes) {
        char* p = ws + off;
        off = (off + bytes + 255) & ~(size_t)255;
        return p;
    };
    float*          xn_f  = (float*)take((size_t)BN * D * sizeof(float));
    unsigned short* xn_bf = (unsigned short*)take((size_t)BN * D * sizeof(unsigned short));
    float*          x2_v  = (float*)take((size_t)BN * sizeof(float));
    unsigned short* cb_bf = (unsigned short*)take((size_t)K * D * sizeof(unsigned short));
    float*          e2_v  = (float*)take((size_t)K * sizeof(float));
    int*            ind_i = (int*)take((size_t)BN * sizeof(int));
    (void)ws_size; (void)n_in; (void)out_size;

    k_layernorm<<<BN / 8, 256, 0, stream>>>(x, xn_f, xn_bf, x2_v);
    k_prepcb<<<K / 8, 256, 0, stream>>>(cb, cb_bf, e2_v);
    k_argmin<<<BN / 128, 256, 0, stream>>>(xn_bf, x2_v, cb_bf, e2_v,
                                           ind_i, out + ind_off, K);
    k_zero<<<1, 64, 0, stream>>>(out + loss_off);
    k_gather<<<BN / 8, 256, 0, stream>>>(xn_f, cb, ind_i, out + q_off,
                                         out + loss_off,
                                         1.0f / ((float)BN * (float)D));
}
